// GAT_67869073212171
// MI455X (gfx1250) — matrix operand order repaired and checked by reading, not compiled, not run on hardware
//
#include <hip/hip_runtime.h>
#include <hip/hip_bf16.h>
#include <math.h>

// ---------------------------------------------------------------------------
// Types (gfx1250, wave32)
// ---------------------------------------------------------------------------
typedef __attribute__((ext_vector_type(16))) __bf16 v16bf;
typedef __attribute__((ext_vector_type(8)))  float  v8f;
typedef __attribute__((ext_vector_type(4)))  unsigned int u32x4;
typedef __attribute__((ext_vector_type(4)))  int i32x4;
typedef __attribute__((ext_vector_type(8)))  int i32x8;

union FragBF { unsigned u[8]; v16bf v; };

__device__ __forceinline__ unsigned short f2bf(float f) {
    unsigned u = __float_as_uint(f);
    unsigned r = 0x7FFFu + ((u >> 16) & 1u);   // round-to-nearest-even
    return (unsigned short)((u + r) >> 16);
}

__device__ __forceinline__ float eluf(float x)  { return x > 0.f ? x : expm1f(x); }
__device__ __forceinline__ float lrelu(float x) { return x > 0.f ? x : 0.2f * x; }

#ifndef HC_
#define HC_ 384
#define H_ 6
#define C_ 64
#define NPG_ 228
#endif

// ---------------------------------------------------------------------------
// Tensor Data Mover: 2D bf16 tile load global->LDS.
//   tensor dims are the REMAINING extent from the tile start (global_addr is
//   the tile start per D# spec); reads past them are zero-filled by TDM.
//   Issued per-wave; tracked by TENSORcnt.  (6-arg clang-23 builtin form.)
// ---------------------------------------------------------------------------
__device__ __forceinline__ void tdm_load_tile_bf16(
    unsigned ldsByteAddr, const unsigned short* gtile,
    unsigned remDim0, unsigned remDim1,
    unsigned tileDim0, unsigned tileDim1,
    unsigned rowStrideElems) {
    unsigned long long ga = (unsigned long long)(const void*)gtile;
    u32x4 g0;
    g0[0] = 1u;                                    // count=1, user mode, no gather
    g0[1] = ldsByteAddr;                           // lds_addr [63:32]
    g0[2] = (unsigned)(ga & 0xFFFFFFFFull);        // global_addr[31:0]
    g0[3] = (unsigned)((ga >> 32) & 0x01FFFFFFull) // global_addr[56:32]
            | (2u << 30);                          // type=2 ("image")
    i32x8 g1;
    g1[0] = (int)(1u << 16);                       // data_size=1 -> 2 bytes
    g1[1] = (int)((remDim0 & 0xFFFFu) << 16);      // tensor_dim0[15:0] at [63:48]
    g1[2] = (int)((remDim0 >> 16) | ((remDim1 & 0xFFFFu) << 16)); // td0 hi | td1 lo
    g1[3] = (int)((remDim1 >> 16) | (tileDim0 << 16));            // td1 hi | tile_dim0
    g1[4] = (int)(tileDim1 & 0xFFFFu);             // tile_dim1 (tile_dim2 = 0)
    g1[5] = (int)rowStrideElems;                   // tensor_dim0_stride[31:0]
    g1[6] = 0;                                     // stride hi | dim1_stride lo
    g1[7] = 0;
    i32x4 z4 = {0, 0, 0, 0};
    i32x8 z8 = {0, 0, 0, 0, 0, 0, 0, 0};
    __builtin_amdgcn_tensor_load_to_lds(g0, g1, z4, z4, z8, 0);
}

__device__ __forceinline__ unsigned lds_off(const void* p) {
    return (unsigned)(unsigned long long)(uintptr_t)p;   // low 32b = LDS byte offset
}

// ---------------------------------------------------------------------------
// Utility fills / conversions
// ---------------------------------------------------------------------------
__global__ void k_zero_u32(unsigned* p, int n) {
    for (int i = blockIdx.x * blockDim.x + threadIdx.x; i < n; i += gridDim.x * blockDim.x)
        p[i] = 0u;
}
__global__ void k_zero_f32(float* p, int n) {
    for (int i = blockIdx.x * blockDim.x + threadIdx.x; i < n; i += gridDim.x * blockDim.x)
        p[i] = 0.f;
}
__global__ void k_cvt_bf16(const float* __restrict__ in, unsigned short* __restrict__ out, int n) {
    for (int i = blockIdx.x * blockDim.x + threadIdx.x; i < n; i += gridDim.x * blockDim.x)
        out[i] = f2bf(in[i]);
}
// out[c][r] = bf16(in[r][c]) : in R x Cn row-major  ->  out Cn x R row-major
__global__ __launch_bounds__(256)
void k_transpose_cvt(const float* __restrict__ in, unsigned short* __restrict__ out,
                     int R, int Cn) {
    __shared__ float tile[32][33];
    const int rb = blockIdx.y * 32, cb = blockIdx.x * 32;
    const int tx = threadIdx.x, ty = threadIdx.y;   // (32, 8)
#pragma unroll
    for (int i = 0; i < 4; ++i) {
        int r = rb + ty + 8 * i, c = cb + tx;
        tile[ty + 8 * i][tx] = (r < R && c < Cn) ? in[(size_t)r * Cn + c] : 0.f;
    }
    __syncthreads();
#pragma unroll
    for (int i = 0; i < 4; ++i) {
        int oc = cb + ty + 8 * i, orr = rb + tx;    // out row = c, out col = r
        if (oc < Cn && orr < R)
            out[(size_t)oc * R + orr] = f2bf(tile[tx][ty + 8 * i]);
    }
}

// ---------------------------------------------------------------------------
// CSR construction: degree count, exclusive scan, scatter (incl. self-loops)
// ---------------------------------------------------------------------------
__global__ void k_degree(const int* dstA, unsigned* deg, int E, int Etot) {
    int i = blockIdx.x * blockDim.x + threadIdx.x;
    if (i >= Etot) return;
    int d = (i < E) ? dstA[i] : (i - E);
    atomicAdd(&deg[d], 1u);
}

__global__ void k_scan(const unsigned* deg, unsigned* rowptr, unsigned* cursor, int n) {
    __shared__ unsigned s[1024];
    __shared__ unsigned carry;
    int t = threadIdx.x;
    if (t == 0) carry = 0u;
    __syncthreads();
    for (int base = 0; base < n; base += 1024) {
        unsigned v = (base + t < n) ? deg[base + t] : 0u;
        s[t] = v;
        __syncthreads();
        for (int off = 1; off < 1024; off <<= 1) {
            unsigned add = (t >= off) ? s[t - off] : 0u;
            __syncthreads();
            s[t] += add;
            __syncthreads();
        }
        unsigned excl = s[t] - v;
        unsigned tot  = s[1023];
        unsigned c    = carry;
        if (base + t < n) { rowptr[base + t] = c + excl; cursor[base + t] = c + excl; }
        __syncthreads();
        if (t == 0) carry = c + tot;
        __syncthreads();
    }
    if (t == 0) rowptr[n] = carry;
}

__global__ void k_scatter(const int* srcA, const int* dstA, unsigned* cursor,
                          unsigned* csr_src, int E, int Etot) {
    int i = blockIdx.x * blockDim.x + threadIdx.x;
    if (i >= Etot) return;
    int s, d;
    if (i < E) { s = srcA[i]; d = dstA[i]; }
    else       { s = i - E;   d = i - E;   }
    unsigned pos = atomicAdd(&cursor[d], 1u);
    csr_src[pos] = (unsigned)s;
}

// ---------------------------------------------------------------------------
// WMMA GEMM with TDM tile staging, double-buffered.
//   C[M x Nn] = A[M x K] @ W[K x Nn]
//   A:  bf16 row-major [M][K];  Bt: bf16 N-major [Nn][K]
//   block tile 64(M) x 128(N), BK=32; 256 threads = 8 waves
// ---------------------------------------------------------------------------
__global__ __launch_bounds__(256)
void k_gemm_tdm(const unsigned short* __restrict__ A, const unsigned short* __restrict__ Bt,
                float* __restrict__ C, int M, int K, int Nn) {
    __shared__ unsigned short As[2][64 * 32];    // [m][k], stride 32
    __shared__ unsigned short Bs[2][128 * 32];   // [n][k], stride 32

    const int tid   = threadIdx.x;
    const int mbase = blockIdx.x * 64;
    const int nbase = blockIdx.y * 128;

    const int wave  = tid >> 5, lane = tid & 31;
    const int mw    = wave & 3, nh = wave >> 2;
    const int hi    = lane >> 4;
    const int mr    = lane & 15;
    const int nlane = lane & 15;

    v8f acc[4];
    const v8f vz = {0.f,0.f,0.f,0.f,0.f,0.f,0.f,0.f};
#pragma unroll
    for (int i = 0; i < 4; ++i) acc[i] = vz;

    const int kTiles = (K + 31) >> 5;

    // prologue: stage tile 0 into buffer 0 (wave 0 owns the TDM queue)
    if (tid < 32) {
        tdm_load_tile_bf16(lds_off(&As[0][0]), A + (size_t)mbase * K,
                           (unsigned)K, (unsigned)(M - mbase), 32u, 64u, (unsigned)K);
        tdm_load_tile_bf16(lds_off(&Bs[0][0]), Bt + (size_t)nbase * K,
                           (unsigned)K, (unsigned)(Nn - nbase), 32u, 128u, (unsigned)K);
    }

    int buf = 0;
    for (int kt = 0; kt < kTiles; ++kt) {
        if (tid < 32) {
            if (kt + 1 < kTiles) {
                const int k1 = (kt + 1) << 5;
                tdm_load_tile_bf16(lds_off(&As[buf ^ 1][0]),
                                   A + (size_t)mbase * K + k1,
                                   (unsigned)(K - k1), (unsigned)(M - mbase),
                                   32u, 64u, (unsigned)K);
                tdm_load_tile_bf16(lds_off(&Bs[buf ^ 1][0]),
                                   Bt + (size_t)nbase * K + k1,
                                   (unsigned)(K - k1), (unsigned)(Nn - nbase),
                                   32u, 128u, (unsigned)K);
                __builtin_amdgcn_s_wait_tensorcnt(2);   // current pair done
            } else {
                __builtin_amdgcn_s_wait_tensorcnt(0);
            }
        }
        __syncthreads();

        // A fragment: lanes 0-15 -> K {0..15}; lanes 16-31 -> +16
        FragBF af;
        const int arow = mw * 16 + mr;
        const int kbA  = hi * 8;
#pragma unroll
        for (int j = 0; j < 8; ++j) {
            int kp = 2 * kbA + 2 * j;
            af.u[j] = *(const unsigned*)&As[buf][arow * 32 + kp];
        }
        const int kbB = hi * 16;
#pragma unroll
        for (int nt = 0; nt < 4; ++nt) {
            FragBF bf_;
            const int bn = nh * 64 + nt * 16 + nlane;
#pragma unroll
            for (int j = 0; j < 8; ++j)
                bf_.u[j] = *(const unsigned*)&Bs[buf][bn * 32 + kbB + 2 * j];
            acc[nt] = __builtin_amdgcn_wmma_f32_16x16x32_bf16(
                false, af.v, false, bf_.v, (short)0, acc[nt], false, false);
        }
        __syncthreads();
        buf ^= 1;
    }

#pragma unroll
    for (int nt = 0; nt < 4; ++nt) {
#pragma unroll
        for (int r = 0; r < 8; ++r) {
            int row = mbase + mw * 16 + r + hi * 8;
            int col = nbase + nh * 64 + nt * 16 + nlane;
            if (row < M) C[(size_t)row * Nn + col] = acc[nt][r];
        }
    }
}

// ---------------------------------------------------------------------------
// Split-K WMMA lin1 with TDM staging: out[48 x 64] += A[48 x K] @ Wt[64 x K]^T
// 128 threads = 4 waves; wave owns 16 cols, all 3 m-tiles. f32 atomic reduce.
// ---------------------------------------------------------------------------
__global__ __launch_bounds__(128)
void k_lin1_tdm(const unsigned short* __restrict__ A, const unsigned short* __restrict__ Bt,
                float* __restrict__ out, int K, int ktilesPerBlk) {
    __shared__ unsigned short As[2][48 * 32];
    __shared__ unsigned short Bs[2][64 * 32];

    const int tid  = threadIdx.x;
    const int wave = tid >> 5, lane = tid & 31;
    const int hi   = lane >> 4;
    const int mr   = lane & 15;
    const int nlane = lane & 15;

    v8f acc[3];
    const v8f vz = {0.f,0.f,0.f,0.f,0.f,0.f,0.f,0.f};
#pragma unroll
    for (int i = 0; i < 3; ++i) acc[i] = vz;

    const int totTiles = (K + 31) >> 5;
    const int ktBase = blockIdx.x * ktilesPerBlk;
    int nTiles = totTiles - ktBase;
    if (nTiles > ktilesPerBlk) nTiles = ktilesPerBlk;
    if (nTiles <= 0) return;

    if (tid < 32) {
        const int k0 = ktBase << 5;
        tdm_load_tile_bf16(lds_off(&As[0][0]), A + k0,
                           (unsigned)(K - k0), 48u, 32u, 48u, (unsigned)K);
        tdm_load_tile_bf16(lds_off(&Bs[0][0]), Bt + k0,
                           (unsigned)(K - k0), 64u, 32u, 64u, (unsigned)K);
    }

    int buf = 0;
    for (int it = 0; it < nTiles; ++it) {
        if (tid < 32) {
            if (it + 1 < nTiles) {
                const int k1 = (ktBase + it + 1) << 5;
                tdm_load_tile_bf16(lds_off(&As[buf ^ 1][0]), A + k1,
                                   (unsigned)(K - k1), 48u, 32u, 48u, (unsigned)K);
                tdm_load_tile_bf16(lds_off(&Bs[buf ^ 1][0]), Bt + k1,
                                   (unsigned)(K - k1), 64u, 32u, 64u, (unsigned)K);
                __builtin_amdgcn_s_wait_tensorcnt(2);
            } else {
                __builtin_amdgcn_s_wait_tensorcnt(0);
            }
        }
        __syncthreads();

        FragBF bf_;
        const int kbB = hi * 16;
        const int bn = wave * 16 + nlane;
#pragma unroll
        for (int j = 0; j < 8; ++j)
            bf_.u[j] = *(const unsigned*)&Bs[buf][bn * 32 + kbB + 2 * j];

        const int kbA = hi * 8;
#pragma unroll
        for (int mt = 0; mt < 3; ++mt) {
            FragBF af;
            const int arow = mt * 16 + mr;
#pragma unroll
            for (int j = 0; j < 8; ++j) {
                int kp = 2 * kbA + 2 * j;
                af.u[j] = *(const unsigned*)&As[buf][arow * 32 + kp];
            }
            acc[mt] = __builtin_amdgcn_wmma_f32_16x16x32_bf16(
                false, af.v, false, bf_.v, (short)0, acc[mt], false, false);
        }
        __syncthreads();
        buf ^= 1;
    }

#pragma unroll
    for (int mt = 0; mt < 3; ++mt) {
#pragma unroll
        for (int r = 0; r < 8; ++r) {
            int row = mt * 16 + r + hi * 8;
            int col = wave * 16 + nlane;
            atomicAdd(&out[row * 64 + col], acc[mt][r]);
        }
    }
}

// ---------------------------------------------------------------------------
// alpha_src / alpha_dst:  a1[n,h] = <h[n,h,:], as[h,:]>,  a2 likewise
// ---------------------------------------------------------------------------
__global__ void k_alpha(const float* __restrict__ h, const float* __restrict__ as_,
                        const float* __restrict__ ad_, float* __restrict__ a1,
                        float* __restrict__ a2, int NH) {
    int i = blockIdx.x * blockDim.x + threadIdx.x;
    if (i >= NH) return;
    int n = i / H_, hh = i - n * H_;
    const float* hp  = h   + (size_t)n * HC_ + hh * C_;
    const float* asp = as_ + hh * C_;
    const float* adp = ad_ + hh * C_;
    float s1 = 0.f, s2 = 0.f;
#pragma unroll 4
    for (int c = 0; c < C_; ++c) {
        float v = hp[c];
        s1 += v * asp[c];
        s2 += v * adp[c];
    }
    a1[i] = s1;
    a2[i] = s2;
}

// ---------------------------------------------------------------------------
// Per-destination-node segment softmax + weighted aggregation (one wave/node)
// ---------------------------------------------------------------------------
__global__ __launch_bounds__(256)
void k_gat_agg(const float* __restrict__ h, const float* __restrict__ a1,
               const float* __restrict__ a2, const unsigned* __restrict__ rowptr,
               const unsigned* __restrict__ csr_src, const float* __restrict__ bias,
               float* __restrict__ out, int N) {
    const int lane = threadIdx.x & 31;
    const int node = blockIdx.x * 8 + (threadIdx.x >> 5);
    if (node >= N) return;

    const unsigned r0 = rowptr[node], r1 = rowptr[node + 1];

    float a2d[H_];
#pragma unroll
    for (int hh = 0; hh < H_; ++hh) a2d[hh] = a2[node * H_ + hh];

    float mh[H_];
#pragma unroll
    for (int hh = 0; hh < H_; ++hh) mh[hh] = -1e30f;
    for (unsigned i = r0 + lane; i < r1; i += 32) {
        int s = (int)csr_src[i];
#pragma unroll
        for (int hh = 0; hh < H_; ++hh) {
            float e = lrelu(a1[s * H_ + hh] + a2d[hh]);
            mh[hh] = fmaxf(mh[hh], e);
        }
    }
#pragma unroll
    for (int off = 16; off > 0; off >>= 1)
#pragma unroll
        for (int hh = 0; hh < H_; ++hh)
            mh[hh] = fmaxf(mh[hh], __shfl_xor(mh[hh], off, 32));

    float accv[12];
#pragma unroll
    for (int j = 0; j < 12; ++j) accv[j] = 0.f;
    float Dh[H_];
#pragma unroll
    for (int hh = 0; hh < H_; ++hh) Dh[hh] = 0.f;

    for (unsigned i = r0; i < r1; ++i) {
        int s = (int)csr_src[i];          // uniform across wave -> broadcast loads
        float wv[H_];
#pragma unroll
        for (int hh = 0; hh < H_; ++hh) {
            float e = lrelu(a1[s * H_ + hh] + a2d[hh]);
            wv[hh] = __expf(e - mh[hh]);
            Dh[hh] += wv[hh];
        }
        const float* hp = h + (size_t)s * HC_;
#pragma unroll
        for (int j = 0; j < 12; ++j) {
            int c = lane + 32 * j;
            accv[j] += wv[c >> 6] * hp[c];
        }
    }

#pragma unroll
    for (int j = 0; j < 12; ++j) {
        int c = lane + 32 * j;
        float o = accv[j] / (Dh[c >> 6] + 1e-16f) + bias[c];
        out[(size_t)node * HC_ + c] = o;
    }
}

// ---------------------------------------------------------------------------
// Fused ELU + GraphNorm, bf16 output (feeds the next WMMA GEMM / lin1).
// block per graph, thread per channel (384 threads)
// ---------------------------------------------------------------------------
__global__ __launch_bounds__(HC_)
void k_elu_gn_bf16(const float* __restrict__ x, unsigned short* __restrict__ y,
                   const float* __restrict__ gamma, const float* __restrict__ beta,
                   const float* __restrict__ ms) {
    const int c = threadIdx.x;
    const int b = blockIdx.x;
    const size_t base = (size_t)b * NPG_ * HC_ + c;

    float s = 0.f;
    for (int n = 0; n < NPG_; ++n) s += eluf(x[base + (size_t)n * HC_]);
    const float mean = s * (1.f / NPG_);
    const float msm  = ms[c] * mean;

    float v = 0.f;
    for (int n = 0; n < NPG_; ++n) {
        float t = eluf(x[base + (size_t)n * HC_]) - msm;
        v += t * t;
    }
    const float inv = rsqrtf(v * (1.f / NPG_) + 1e-5f);
    const float g = gamma[c], bt = beta[c];

    for (int n = 0; n < NPG_; ++n) {
        float t = (eluf(x[base + (size_t)n * HC_]) - msm) * inv;
        y[base + (size_t)n * HC_] = f2bf(g * t + bt);
    }
}

// ---------------------------------------------------------------------------
// Head: elu(lin1+b) -> BN(eval) -> lin2. Block per graph, 64 threads.
// ---------------------------------------------------------------------------
__global__ __launch_bounds__(64)
void k_head(const float* __restrict__ lacc, const float* __restrict__ l1b,
            const float* __restrict__ bw, const float* __restrict__ bb,
            const float* __restrict__ rm, const float* __restrict__ rv,
            const float* __restrict__ l2w, const float* __restrict__ l2b,
            float* __restrict__ out) {
    __shared__ float y[64];
    const int j = threadIdx.x, b = blockIdx.x;
    float t = lacc[b * 64 + j] + l1b[j];
    t = eluf(t);
    t = (t - rm[j]) * rsqrtf(rv[j] + 1e-5f) * bw[j] + bb[j];
    y[j] = t;
    __syncthreads();
    if (j < 2) {
        float s = l2b[j];
#pragma unroll 8
        for (int k = 0; k < 64; ++k) s += y[k] * l2w[k * 2 + j];
        out[b * 2 + j] = s;
    }
}

// ---------------------------------------------------------------------------
// Launcher
// ---------------------------------------------------------------------------
extern "C" void kernel_launch(void* const* d_in, const int* in_sizes, int n_in,
                              void* d_out, int out_size, void* d_ws, size_t ws_size,
                              hipStream_t stream) {
    const float* x    = (const float*)d_in[0];
    const int*   ei   = (const int*)d_in[1];
    const float* w1   = (const float*)d_in[3];
    const float* as1  = (const float*)d_in[4];
    const float* ad1  = (const float*)d_in[5];
    const float* b1   = (const float*)d_in[6];
    const float* w2   = (const float*)d_in[7];
    const float* as2  = (const float*)d_in[8];
    const float* ad2  = (const float*)d_in[9];
    const float* b2   = (const float*)d_in[10];
    const float* gn1w = (const float*)d_in[11];
    const float* gn1b = (const float*)d_in[12];
    const float* gn1m = (const float*)d_in[13];
    const float* gn2w = (const float*)d_in[14];
    const float* gn2b = (const float*)d_in[15];
    const float* gn2m = (const float*)d_in[16];
    const float* l1w  = (const float*)d_in[17];
    const float* l1b  = (const float*)d_in[18];
    const float* bnw  = (const float*)d_in[19];
    const float* bnb  = (const float*)d_in[20];
    const float* bnrm = (const float*)d_in[21];
    const float* bnrv = (const float*)d_in[22];
    const float* l2w  = (const float*)d_in[23];
    const float* l2b  = (const float*)d_in[24];
    float* out = (float*)d_out;

    const int N    = in_sizes[2];          // 10944
    const int E    = in_sizes[1] / 2;      // 350208
    const int Bg   = N / NPG_;             // 48
    const int F    = in_sizes[0] / N;      // 228
    const int Etot = E + N;
    const int Kflat = NPG_ * HC_;          // 87552

    const int* srcA = ei;
    const int* dstA = ei + E;

    // workspace carve-out (256B aligned)
    char* w = (char*)d_ws;
    auto carve = [&](size_t bytes) { char* p = w; w += (bytes + 255) & ~(size_t)255; return p; };
    float*          hbuf  = (float*)         carve((size_t)N * HC_ * 4);   // GEMM out (pre-agg h)
    float*          gbuf  = (float*)         carve((size_t)N * HC_ * 4);   // agg out
    unsigned short* gnbf  = (unsigned short*)carve((size_t)N * HC_ * 2);   // GN out (bf16)
    unsigned short* xbf   = (unsigned short*)carve((size_t)N * F * 2);     // x (bf16)
    unsigned short* w1t   = (unsigned short*)carve((size_t)HC_ * F * 2);   // w1^T (bf16)
    unsigned short* w2t   = (unsigned short*)carve((size_t)HC_ * HC_ * 2); // w2^T (bf16)
    unsigned short* l1wt  = (unsigned short*)carve((size_t)64 * Kflat * 2);// lin1_w^T (bf16)
    float*          a1    = (float*)         carve((size_t)N * H_ * 4);
    float*          a2    = (float*)         carve((size_t)N * H_ * 4);
    unsigned*       deg   = (unsigned*)      carve((size_t)N * 4);
    unsigned*       rowptr= (unsigned*)      carve((size_t)(N + 1) * 4);
    unsigned*       cursor= (unsigned*)      carve((size_t)N * 4);
    unsigned*       csr   = (unsigned*)      carve((size_t)Etot * 4);
    float*          lacc  = (float*)         carve((size_t)Bg * 64 * 4);

    // --- prep: bf16 conversions / weight transposes ---
    k_cvt_bf16<<<(N * F + 255) / 256, 256, 0, stream>>>(x, xbf, N * F);
    {
        dim3 blk(32, 8);
        dim3 g1((HC_ + 31) / 32, (F + 31) / 32);          // w1: F x HC -> HC x F
        k_transpose_cvt<<<g1, blk, 0, stream>>>(w1, w1t, F, HC_);
        dim3 g2((HC_ + 31) / 32, (HC_ + 31) / 32);        // w2
        k_transpose_cvt<<<g2, blk, 0, stream>>>(w2, w2t, HC_, HC_);
        dim3 g3((64 + 31) / 32, (Kflat + 31) / 32);       // lin1_w: Kflat x 64 -> 64 x Kflat
        k_transpose_cvt<<<g3, blk, 0, stream>>>(l1w, l1wt, Kflat, 64);
    }

    // --- CSR build (shared by both GAT layers) ---
    k_zero_u32<<<(N + 255) / 256, 256, 0, stream>>>(deg, N);
    k_degree<<<(Etot + 255) / 256, 256, 0, stream>>>(dstA, deg, E, Etot);
    k_scan<<<1, 1024, 0, stream>>>(deg, rowptr, cursor, N);
    k_scatter<<<(Etot + 255) / 256, 256, 0, stream>>>(srcA, dstA, cursor, csr, E, Etot);

    const dim3 gemmGrid((N + 63) / 64, HC_ / 128);
    const int NH = N * H_;

    // --- GAT layer 1 ---
    k_gemm_tdm<<<gemmGrid, 256, 0, stream>>>(xbf, w1t, hbuf, N, F, HC_);
    k_alpha<<<(NH + 255) / 256, 256, 0, stream>>>(hbuf, as1, ad1, a1, a2, NH);
    k_gat_agg<<<(N + 7) / 8, 256, 0, stream>>>(hbuf, a1, a2, rowptr, csr, b1, gbuf, N);
    k_elu_gn_bf16<<<Bg, HC_, 0, stream>>>(gbuf, gnbf, gn1w, gn1b, gn1m);

    // --- GAT layer 2 ---
    k_gemm_tdm<<<gemmGrid, 256, 0, stream>>>(gnbf, w2t, hbuf, N, HC_, HC_);
    k_alpha<<<(NH + 255) / 256, 256, 0, stream>>>(hbuf, as2, ad2, a1, a2, NH);
    k_gat_agg<<<(N + 7) / 8, 256, 0, stream>>>(hbuf, a1, a2, rowptr, csr, b2, gbuf, N);
    k_elu_gn_bf16<<<Bg, HC_, 0, stream>>>(gbuf, gnbf, gn2w, gn2b, gn2m);

    // --- lin1 (split-K WMMA, TDM staging, f32 atomic reduce) ---
    k_zero_f32<<<(Bg * 64 + 255) / 256, 256, 0, stream>>>(lacc, Bg * 64);
    const int totTiles = (Kflat + 31) / 32;      // 2736
    const int nBlocks  = 114;
    const int ktPerBlk = (totTiles + nBlocks - 1) / nBlocks;  // 24
    k_lin1_tdm<<<nBlocks, 128, 0, stream>>>(gnbf, l1wt, lacc, Kflat, ktPerBlk);

    // --- head ---
    k_head<<<Bg, 64, 0, stream>>>(lacc, l1b, bnw, bnb, bnrm, bnrv, l2w, l2b, out);
}
